// GAT_Graph_68350109549110
// MI455X (gfx1250) — compile-verified, matrix operand order checked by reading, not run on hardware
//
#include <hip/hip_runtime.h>
#include <hip/hip_bf16.h>
#include <math.h>

// ---------------- output layout (float elements) ----------------
#define OFF_UHA 0        // user_history_aspect_feature (4,21,1,10,100) = 84000
#define OFF_PHA 84000    // poi_history_aspect_feature  (4,21,10,100)   = 84000
#define OFF_IUF 168000   // input_user_feature (4,128)                  = 512
#define OFF_PHF 168512   // poi_history_feature (4,21,128)              = 10752
#define OFF_UHF 179264   // user_history_feature (4,21,1,128)           = 10752

typedef __attribute__((ext_vector_type(16))) __bf16       v16bf;
typedef __attribute__((ext_vector_type(8)))  float        v8f;
typedef __attribute__((ext_vector_type(8)))  unsigned int v8u;

// Load one 16x32 bf16 WMMA fragment from a row-major LDS tile.
// Caller passes rowbase = tile + row*ld ; layout per CDNA5 ISA 7.12.2:
// lanes 0-15 hold K 0..7 & 16..23, lanes 16-31 hold K 8..15 & 24..31,
// two bf16 per VGPR -> 8 dword loads (compiler pairs them into ds_load_2addr).
__device__ __forceinline__ v16bf frag_ld(const __bf16* rowbase, int k0, int lane) {
  const unsigned int* p = (const unsigned int*)(rowbase + k0);
  const int h4 = (lane >> 4) << 2;
  v8u u;
#pragma unroll
  for (int i = 0; i < 4; ++i) u[i] = p[i + h4];
#pragma unroll
  for (int i = 0; i < 4; ++i) u[4 + i] = p[8 + i + h4];
  return __builtin_bit_cast(v16bf, u);
}

// ---------------- GAT over word graphs (uh and ph layers) ----------------
// One workgroup per graph: 64 nodes (63 words + prior), d_in=200, d_out=100, 4 heads.
// Only row 63 of (attn @ hp) is needed by the reference.
#define LDH  228   // sHIN [64][228] bf16 (K padded 200->224, +4 anti-bank-conflict pad)
#define LDW  228   // sWT  [112][228] bf16, W_lin transposed [n][k]
#define LDH2 132   // sH2  [64][132] bf16 (K padded 100->128)
#define LDW2 132   // per-head transposed w_head [112][132]
#define LDHP 113   // sHP  [64][113] f32

#define SM_HIN 0
#define SM_WT  29184
#define SM_H2  80256
#define SM_HP  97152
#define SM_D   126080
#define SM_E   126336
#define SM_O   126592
#define SM_S   127040
#define SM_TOTAL 127056

__global__ __launch_bounds__(256) void gat_words_kernel(
    const float* __restrict__ x,            // [G,63,200]
    const float* __restrict__ prior,        // [G,200]
    const unsigned char* __restrict__ mask, // [G,64] (bool)
    const float* __restrict__ Wlin,         // [200,100]
    const float* __restrict__ whead,        // [4,100,100]
    const float* __restrict__ asrc,         // [4,100]
    const float* __restrict__ adst,         // [4,100]
    const float* __restrict__ bias,         // [100]
    float* __restrict__ outAspect)          // [G,100]
{
  extern __shared__ unsigned char smem[];
  __bf16* sHIN = (__bf16*)(smem + SM_HIN);
  __bf16* sWT  = (__bf16*)(smem + SM_WT);
  __bf16* sH2  = (__bf16*)(smem + SM_H2);
  float*  sHP  = (float*)(smem + SM_HP);
  float*  sD   = (float*)(smem + SM_D);
  float*  sE   = (float*)(smem + SM_E);
  float*  sOut = (float*)(smem + SM_O);
  float*  sScal= (float*)(smem + SM_S);

  const int g    = blockIdx.x;
  const int tid  = threadIdx.x;
  const int lane = tid & 31;
  const int wave = tid >> 5;

  const float* xg = x + (size_t)g * (63 * 200);
  const float* pg = prior + (size_t)g * 200;

  // Stage h_in = concat(x, prior) as bf16, zero-padded to 64x224.
  for (int idx = tid; idx < 64 * LDH; idx += 256) {
    int r = idx / LDH, c = idx % LDH;
    float v = 0.f;
    if (c < 200) v = (r < 63) ? xg[r * 200 + c] : pg[c];
    sHIN[idx] = (__bf16)v;
  }
  // Stage W_lin transposed [n][k], zero-padded to 112x224.
  for (int idx = tid; idx < 112 * LDW; idx += 256) {
    int n = idx / LDW, k = idx % LDW;
    float v = (n < 100 && k < 200) ? Wlin[k * 100 + n] : 0.f;
    sWT[idx] = (__bf16)v;
  }
  for (int idx = tid; idx < 64 * LDH2; idx += 256) sH2[idx] = (__bf16)0.f;
  if (tid < 112) sOut[tid] = 0.f;
  __syncthreads();

  // GEMM1: h2(64x100) = h_in(64x224) @ W_lin(224x112), bf16 WMMA, f32 accum.
  // Software-pipelined: issue k+1 fragment loads before consuming k in WMMA so
  // the s_wait before each WMMA only covers the in-flight next-step loads.
  for (int t = wave; t < 28; t += 8) {
    const int mt = t / 7, nt = t % 7;
    const __bf16* arow = sHIN + (mt * 16 + (lane & 15)) * LDH;
    const __bf16* brow = sWT  + (nt * 16 + (lane & 15)) * LDW;
    v8f acc = {};
    v16bf a = frag_ld(arow, 0, lane);
    v16bf b = frag_ld(brow, 0, lane);
#pragma unroll
    for (int kk = 32; kk < 224; kk += 32) {
      v16bf an = frag_ld(arow, kk, lane);
      v16bf bn = frag_ld(brow, kk, lane);
      acc = __builtin_amdgcn_wmma_f32_16x16x32_bf16(false, a, false, b,
                                                    (short)0, acc, false, false);
      a = an; b = bn;
    }
    acc = __builtin_amdgcn_wmma_f32_16x16x32_bf16(false, a, false, b,
                                                  (short)0, acc, false, false);
    const int m0 = mt * 16 + ((lane >> 4) << 3);
    const int n0 = nt * 16 + (lane & 15);
#pragma unroll
    for (int r = 0; r < 8; ++r) sH2[(m0 + r) * LDH2 + n0] = (__bf16)acc[r];
  }
  __syncthreads();

  const bool m63 = mask[(size_t)g * 64 + 63] != 0;
  const bool mj  = (tid < 64) ? (mask[(size_t)g * 64 + tid] != 0) : false;

  for (int h = 0; h < 4; ++h) {
    // Stage per-head weights transposed [e][d], zero-padded 112x128 (reuses sWT).
    for (int idx = tid; idx < 112 * LDW2; idx += 256) {
      int n = idx / LDW2, k = idx % LDW2;
      float v = (n < 100 && k < 100) ? whead[h * 10000 + k * 100 + n] : 0.f;
      sWT[idx] = (__bf16)v;
    }
    __syncthreads();

    // GEMM2: hp(64x100) = h2(64x128) @ w_head(128x112), pipelined as above.
    for (int t = wave; t < 28; t += 8) {
      const int mt = t / 7, nt = t % 7;
      const __bf16* arow = sH2 + (mt * 16 + (lane & 15)) * LDH2;
      const __bf16* brow = sWT + (nt * 16 + (lane & 15)) * LDW2;
      v8f acc = {};
      v16bf a = frag_ld(arow, 0, lane);
      v16bf b = frag_ld(brow, 0, lane);
#pragma unroll
      for (int kk = 32; kk < 128; kk += 32) {
        v16bf an = frag_ld(arow, kk, lane);
        v16bf bn = frag_ld(brow, kk, lane);
        acc = __builtin_amdgcn_wmma_f32_16x16x32_bf16(false, a, false, b,
                                                      (short)0, acc, false, false);
        a = an; b = bn;
      }
      acc = __builtin_amdgcn_wmma_f32_16x16x32_bf16(false, a, false, b,
                                                    (short)0, acc, false, false);
      const int m0 = mt * 16 + ((lane >> 4) << 3);
      const int n0 = nt * 16 + (lane & 15);
#pragma unroll
      for (int r = 0; r < 8; ++r) sHP[(m0 + r) * LDHP + n0] = acc[r];
    }
    __syncthreads();

    // Attention scores: s_dst for all nodes, s_src only for node 63.
    if (tid < 64) {
      float acc = 0.f;
      const float* ad = adst + h * 100;
      for (int d = 0; d < 100; ++d) acc += tanhf(sHP[tid * LDHP + d]) * ad[d];
      sD[tid] = acc;
    } else if (tid == 64) {
      float acc = 0.f;
      const float* as = asrc + h * 100;
      for (int d = 0; d < 100; ++d) acc += tanhf(sHP[63 * LDHP + d]) * as[d];
      sScal[0] = acc;
    }
    __syncthreads();
    if (tid < 64) {
      float raw = sScal[0] + sD[tid];
      float l = raw > 0.f ? raw : 0.2f * raw;   // leaky_relu(0.2)
      if (m63 || mj) l = -1e18f;
      sE[tid] = l;
    }
    __syncthreads();
    if (tid == 0) {
      float m = sE[0];
      for (int j = 1; j < 64; ++j) m = fmaxf(m, sE[j]);
      sScal[1] = m;
    }
    __syncthreads();
    if (tid < 64) sE[tid] = expf(sE[tid] - sScal[1]);
    __syncthreads();
    if (tid == 0) {
      float s = 0.f;
      for (int j = 0; j < 64; ++j) s += sE[j];
      sScal[2] = 1.f / s;
    }
    __syncthreads();
    if (tid < 100) {
      float acc = 0.f;
      for (int j = 0; j < 64; ++j) acc += sE[j] * sHP[j * LDHP + tid];
      sOut[tid] += acc * sScal[2];
    }
    __syncthreads();
  }

  if (tid < 100)
    outAspect[(size_t)g * 100 + tid] = sOut[tid] * 0.25f + bias[tid];
}

// ---------------- aspect merge (1000 -> 128) + positional add ----------------
__global__ __launch_bounds__(128) void aspect_merge_kernel(
    const float* __restrict__ Wuser,   // (1000,128)
    const float* __restrict__ Wpoi,    // (1000,128)
    const int*   __restrict__ flag,    // only_use_user_history
    float* __restrict__ dout)
{
  __shared__ float sIn[1000];
  const int r = blockIdx.x;            // 0..83 user rows, 84..167 poi rows
  const int e = threadIdx.x;           // 0..127
  const bool isPoi = (r >= 84);
  const int rr = isPoi ? (r - 84) : r;
  const float* in = dout + (isPoi ? (OFF_PHA + rr * 1000) : (OFF_UHA + rr * 1000));
  const float* W  = isPoi ? Wpoi : Wuser;
  for (int k = e; k < 1000; k += 128) sIn[k] = in[k];
  __syncthreads();
  float acc = 0.f;
  for (int k = 0; k < 1000; ++k) acc += sIn[k] * W[k * 128 + e];  // coalesced over e
  if (isPoi) {
    const int t = rr % 21;
    if ((*flag == 0) && t < 20) {      // sinusoidal positional encoding on [:, :-1, :]
      const int i2 = (e >> 1) << 1;
      const float dv = expf((float)i2 * (-logf(10000.f) / 128.f));
      const float ang = (float)t * dv;
      acc += (e & 1) ? cosf(ang) : sinf(ang);
    }
    dout[OFF_PHF + rr * 128 + e] = acc;
  } else {
    dout[OFF_UHF + rr * 128 + e] = acc;
  }
}

// ---------------- pu GAT (tiny: 4 batches x 21 nodes x 128) ----------------
__global__ __launch_bounds__(256) void pu_gat_kernel(
    const float* __restrict__ priorU,  // (4,128)
    const float* __restrict__ Wlin,    // (128,128)
    const float* __restrict__ whead,   // (4,128,128)
    const float* __restrict__ asrc,    // (4,128)
    const float* __restrict__ adst,    // (4,128)
    const float* __restrict__ bias,    // (128,)
    const unsigned char* __restrict__ mask, // (4,21)
    const int*   __restrict__ flag,
    float* __restrict__ dout)
{
  const int b = blockIdx.x;
  const int tid = threadIdx.x;
  if (*flag != 0) {                    // only_use_user_history path
    if (tid < 128)
      dout[OFF_IUF + b * 128 + tid] = dout[OFF_UHF + (b * 21 + 20) * 128 + tid];
    return;
  }
  __shared__ float sH[21 * 128];
  __shared__ float sHp[21 * 128];
  __shared__ float sD[21];
  __shared__ float sE[21];
  __shared__ float sOut[128];
  __shared__ float sScal[4];

  // prior_at_input = False: h = x @ W_pu_lin for rows 0..19, row 20 = raw prior.
  const float* xb = dout + OFF_PHF + b * 21 * 128;   // poi_history_feature[b]
  for (int idx = tid; idx < 20 * 128; idx += 256) {
    int i = idx >> 7, o = idx & 127;
    float acc = 0.f;
    for (int k = 0; k < 128; ++k) acc += xb[i * 128 + k] * Wlin[k * 128 + o];
    sH[idx] = acc;
  }
  if (tid < 128) { sH[20 * 128 + tid] = priorU[b * 128 + tid]; sOut[tid] = 0.f; }
  __syncthreads();

  const bool m20 = mask[b * 21 + 20] != 0;
  const bool mj  = (tid < 21) ? (mask[b * 21 + tid] != 0) : false;

  for (int h = 0; h < 4; ++h) {
    const float* Wh = whead + h * 128 * 128;
    for (int idx = tid; idx < 21 * 128; idx += 256) {
      int i = idx >> 7, o = idx & 127;
      float acc = 0.f;
      for (int k = 0; k < 128; ++k) acc += sH[i * 128 + k] * Wh[k * 128 + o];
      sHp[idx] = acc;
    }
    __syncthreads();
    if (tid < 21) {
      float acc = 0.f;
      const float* ad = adst + h * 128;
      for (int d = 0; d < 128; ++d) acc += tanhf(sHp[tid * 128 + d]) * ad[d];
      sD[tid] = acc;
    } else if (tid == 21) {
      float acc = 0.f;
      const float* as = asrc + h * 128;
      for (int d = 0; d < 128; ++d) acc += tanhf(sHp[20 * 128 + d]) * as[d];
      sScal[0] = acc;
    }
    __syncthreads();
    if (tid < 21) {
      float raw = sScal[0] + sD[tid];
      float l = raw > 0.f ? raw : 0.2f * raw;
      if (m20 || mj) l = -1e18f;
      sE[tid] = l;
    }
    __syncthreads();
    if (tid == 0) {
      float m = sE[0];
      for (int j = 1; j < 21; ++j) m = fmaxf(m, sE[j]);
      float s = 0.f;
      for (int j = 0; j < 21; ++j) { sE[j] = expf(sE[j] - m); s += sE[j]; }
      sScal[1] = 1.f / s;
    }
    __syncthreads();
    if (tid < 128) {
      float acc = 0.f;
      for (int j = 0; j < 21; ++j) acc += sE[j] * sHp[j * 128 + tid];
      sOut[tid] += acc * sScal[1];
    }
    __syncthreads();
  }
  if (tid < 128)
    dout[OFF_IUF + b * 128 + tid] = sOut[tid] * 0.25f + bias[tid];
}

// ---------------- launch ----------------
extern "C" void kernel_launch(void* const* d_in, const int* in_sizes, int n_in,
                              void* d_out, int out_size, void* d_ws, size_t ws_size,
                              hipStream_t stream) {
  (void)in_sizes; (void)n_in; (void)out_size; (void)d_ws; (void)ws_size;
  float* dout = (float*)d_out;

  hipFuncSetAttribute(reinterpret_cast<const void*>(gat_words_kernel),
                      hipFuncAttributeMaxDynamicSharedMemorySize, SM_TOTAL);

  // uh layer: 840 graphs from (4,21,1,10)
  gat_words_kernel<<<840, 256, SM_TOTAL, stream>>>(
      (const float*)d_in[0], (const float*)d_in[1], (const unsigned char*)d_in[22],
      (const float*)d_in[5], (const float*)d_in[6], (const float*)d_in[7],
      (const float*)d_in[8], (const float*)d_in[9], dout + OFF_UHA);

  // ph layer: 840 graphs from (4,21,10)
  gat_words_kernel<<<840, 256, SM_TOTAL, stream>>>(
      (const float*)d_in[2], (const float*)d_in[3], (const unsigned char*)d_in[23],
      (const float*)d_in[10], (const float*)d_in[11], (const float*)d_in[12],
      (const float*)d_in[13], (const float*)d_in[14], dout + OFF_PHA);

  // aspect merges (reads regions 1&2 of dout) + positional add
  aspect_merge_kernel<<<168, 128, 0, stream>>>(
      (const float*)d_in[20], (const float*)d_in[21], (const int*)d_in[25], dout);

  // pu GAT (reads poi_history_feature / user_history_feature regions of dout)
  pu_gat_kernel<<<4, 256, 0, stream>>>(
      (const float*)d_in[4], (const float*)d_in[15], (const float*)d_in[16],
      (const float*)d_in[17], (const float*)d_in[18], (const float*)d_in[19],
      (const unsigned char*)d_in[24], (const int*)d_in[25], dout);
}
